// EventGenerator_74105365725602
// MI455X (gfx1250) — compile-verified, matrix operand order checked by reading, not run on hardware
//
#include <hip/hip_runtime.h>
#include <math.h>

#define LATENT   256
#define NEVENTS  32
#define NFRAMES  256
#define WINDOW   2048
#define STEP     1024
#define NCOEFFS  1025
#define NSAMPLES 262144
#define BEDIM    64          // B * N_EVENTS * EXPR
#define MGEMM    16384       // BEDIM * NFRAMES
#define KGEMM    2080        // 2*1025 padded up to multiple of 32
#define NGEMM    2048        // WINDOW
#define FFT_N    524288      // 2 * NSAMPLES
#define CHUNK    8           // events per conv chunk
#define SIGS     24          // 3 signals per event (res0, res1, noise)

typedef __attribute__((ext_vector_type(16))) __bf16 bf16x16;
typedef __attribute__((ext_vector_type(8)))  float  floatx8;

__device__ __forceinline__ unsigned short f2bf(float f) {
  unsigned int u = __float_as_uint(f);
  u += 0x7fffu + ((u >> 16) & 1u);          // round-to-nearest-even
  return (unsigned short)(u >> 16);
}
__device__ __forceinline__ float2 cmul(float2 a, float2 b) {
  return make_float2(a.x * b.x - a.y * b.y, a.x * b.y + a.y * b.x);
}

// ---------------------------------------------------------------- stage 1
// Per-event small MLPs: hN, hD, hC, hP, hDef, mix softmax, noise envelope^2
__global__ __launch_bounds__(256) void k_small(
    const float* __restrict__ ev,
    const float* Wn, const float* bn, const float* items_n,
    const float* Wd, const float* bd,
    const float* Wc, const float* bc,
    const float* Wp, const float* bp,
    const float* Wdef, const float* bdef,
    const float* Wmix, const float* bmix,
    float* hN, float* hD, float* hC, float* hP, float* hDef,
    float* wmix, float* env2) {
  __shared__ float se[LATENT];
  __shared__ float shn[32];
  __shared__ float smix[2];
  int ne = blockIdx.x, tid = threadIdx.x;
  se[tid] = ev[ne * LATENT + tid];
  __syncthreads();
  if (tid < 32) {
    float s = bn[tid];
    for (int i = 0; i < LATENT; ++i) s += se[i] * Wn[i * 32 + tid];
    s = fmaxf(s, 0.f); hN[ne * 32 + tid] = s; shn[tid] = s;
  } else if (tid < 64) {
    int o = tid - 32; float s = bd[o];
    for (int i = 0; i < LATENT; ++i) s += se[i] * Wd[i * 32 + o];
    hD[ne * 32 + o] = fmaxf(s, 0.f);
  } else if (tid < 192) {
    int o = tid - 64; float s = bc[o];
    for (int i = 0; i < LATENT; ++i) s += se[i] * Wc[i * 128 + o];
    hC[ne * 128 + o] = fmaxf(s, 0.f);
  } else if (tid < 208) {
    int o = tid - 192; float s = bp[o];
    for (int i = 0; i < LATENT; ++i) s += se[i] * Wp[i * 16 + o];
    hP[ne * 16 + o] = fmaxf(s, 0.f);
  } else if (tid < 240) {
    int o = tid - 208; float s = bdef[o];
    for (int i = 0; i < LATENT; ++i) s += se[i] * Wdef[i * 32 + o];
    hDef[ne * 32 + o] = fmaxf(s, 0.f);
  } else if (tid < 242) {
    int o = tid - 240; float s = bmix[o];
    for (int i = 0; i < LATENT; ++i) s += se[i] * Wmix[i * 2 + o];
    smix[o] = s;
  }
  __syncthreads();
  if (tid == 0) {
    float m = fmaxf(smix[0], smix[1]);
    float e0 = expf(smix[0] - m), e1 = expf(smix[1] - m);
    float inv = 1.f / (e0 + e1);
    wmix[ne * 2 + 0] = e0 * inv; wmix[ne * 2 + 1] = e1 * inv;
  }
  if (tid < 64) {
    float s = 0.f;
    for (int i = 0; i < 32; ++i) s += shn[i] * items_n[i * 64 + tid];
    env2[ne * 64 + tid] = s * s;
  }
}

// ---------------------------------------------------------------- stage 2
__global__ void k_cumdn(const float* __restrict__ dn, float* __restrict__ cumdn) {
  int c = blockIdx.x * 256 + threadIdx.x;
  if (c >= NCOEFFS) return;
  float acc = 0.f;
  for (int f = 0; f < NFRAMES; ++f) {
    acc += dn[f * NCOEFFS + c];
    cumdn[f * NCOEFFS + c] = acc;
  }
}

// per-(be,c) resonance parameters: log(coeff), start_phase, start_mags, tanh(dither)
__global__ void k_params(const float* __restrict__ ev,
                         const float* __restrict__ Wamp, const float* __restrict__ bamp,
                         const float* __restrict__ items_d, const float* __restrict__ items_c,
                         const float* __restrict__ items_p,
                         const float* __restrict__ hD, const float* __restrict__ hC,
                         const float* __restrict__ hP,
                         float* logc, float* sp, float* sm, float* pd) {
  int idx = blockIdx.x * 256 + threadIdx.x;
  if (idx >= BEDIM * NCOEFFS) return;
  int be = idx / NCOEFFS, c = idx % NCOEFFS;
  int ne = be >> 1, e = be & 1;
  int col = e * NCOEFFS + c;
  const float* er = ev + ne * LATENT;
  float amp = bamp[col];
  for (int i = 0; i < LATENT; ++i) amp += er[i] * Wamp[i * 2050 + col];
  float dith = 0.f;
  for (int i = 0; i < 16; ++i) dith += hD[ne * 32 + e * 16 + i] * items_d[i * NCOEFFS + c];
  float dec = 0.f;
  for (int i = 0; i < 64; ++i) dec += hC[ne * 128 + e * 64 + i] * items_c[i * NCOEFFS + c];
  float pha = 0.f;
  for (int i = 0; i < 8; ++i) pha += hP[ne * 16 + e * 8 + i] * items_p[i * NCOEFFS + c];
  float coeff = (1.f / (1.f + expf(-dec))) * 0.9999f;
  logc[idx] = logf(coeff);
  sp[idx] = tanhf(pha) * 3.14159265358979323846f;
  sm[idx] = amp * amp;
  pd[idx] = tanhf(dith);
}

// deformation: relu(cumsum(hDef @ items_def))
__global__ __launch_bounds__(256) void k_defo(const float* __restrict__ hDef,
                                              const float* __restrict__ items_def,
                                              float* __restrict__ dc) {
  __shared__ float sd[NFRAMES];
  int be = blockIdx.x, f = threadIdx.x;
  int ne = be >> 1, e = be & 1;
  float v = 0.f;
  for (int i = 0; i < 16; ++i) v += hDef[ne * 32 + e * 16 + i] * items_def[i * NFRAMES + f];
  sd[f] = v;
  __syncthreads();
  float acc = 0.f;
  for (int ff = 0; ff <= f; ++ff) acc += sd[ff];
  dc[be * NFRAMES + f] = fmaxf(acc, 0.f);
}

// ---------------------------------------------------------------- stage 3: irfft basis (bf16, K-major)
__global__ void k_basis(unsigned short* __restrict__ Bt) {
  long idx = (long)blockIdx.x * 256 + threadIdx.x;
  if (idx >= (long)NGEMM * KGEMM) return;
  int n = (int)(idx / KGEMM), k = (int)(idx % KGEMM);
  float v = 0.f;
  if (k < 2 * NCOEFFS) {
    int c = (k < NCOEFFS) ? k : (k - NCOEFFS);
    float a = (c == 0 || c == 1024) ? 1.f : 2.f;
    int tw = (c * n) & (WINDOW - 1);                   // exact angle reduction
    float ang = (float)tw * (6.28318530717958647692f / 2048.f);
    float si, co; __sincosf(ang, &si, &co);
    v = (k < NCOEFFS) ? (a * co * (1.f / 2048.f)) : (-a * si * (1.f / 2048.f));
  }
  Bt[idx] = f2bf(v);
}

// spectra -> A matrix (bf16): [Re | Im | 0-pad], row m = be*256 + f
__global__ void k_spec(const float* __restrict__ logc, const float* __restrict__ sp,
                       const float* __restrict__ sm, const float* __restrict__ pd,
                       const float* __restrict__ cumdn, unsigned short* __restrict__ A) {
  const int CW = 1056;
  long idx = (long)blockIdx.x * 256 + threadIdx.x;
  if (idx >= (long)MGEMM * CW) return;
  int m = (int)(idx / CW), c = (int)(idx % CW);
  if (c >= NCOEFFS) {
    int z = c - NCOEFFS;
    if (z < KGEMM - 2 * NCOEFFS) A[(long)m * KGEMM + 2 * NCOEFFS + z] = 0;
    return;
  }
  int be = m >> 8, f = m & 255;
  int pix = be * NCOEFFS + c;
  float t = (float)(f + 1);
  float mag = sm[pix] * expf(t * logc[pix]);
  float gd = 3.14159265358979323846f * (float)c * (1.f / 1024.f);
  float ph = gd * (t + pd[pix] * cumdn[f * NCOEFFS + c]) + sp[pix];
  float si, co;
  sincosf(ph, &si, &co);
  A[(long)m * KGEMM + c] = f2bf(mag * co);
  A[(long)m * KGEMM + NCOEFFS + c] = f2bf(mag * si);
}

// ---------------------------------------------------------------- stage 4: WMMA GEMM (irfft) + Hann window
// C[16384 x 2048] = A[16384 x 2080] * Bt^T,  Bt stored [N][K]
__global__ __launch_bounds__(256) void k_gemm(const unsigned short* __restrict__ A,
                                              const unsigned short* __restrict__ Bt,
                                              float* __restrict__ frames) {
  __shared__ unsigned short sA[128][40];   // 32 K-entries + 16B pad per row
  __shared__ unsigned short sB[64][40];
  int tid = threadIdx.x;
  int lane = tid & 31, wave = tid >> 5;
  int bm = blockIdx.x * 128, bnn = blockIdx.y * 64;
  int wm = (wave >> 1) * 32, wn = (wave & 1) * 32;
  int hsel = lane >> 4, rrow = lane & 15;
  floatx8 acc[2][2] = {};
  for (int k0 = 0; k0 < KGEMM; k0 += 32) {
    {
      int ch = tid;
#pragma unroll
      for (int rep = 0; rep < 2; ++rep, ch += 256) {   // A tile: 128 rows x 32 bf16
        int row = ch >> 2, seg = ch & 3;
        const uint4* gp = (const uint4*)(A + (long)(bm + row) * KGEMM + k0 + seg * 8);
        *((uint4*)&sA[row][seg * 8]) = *gp;
      }
      int row = tid >> 2, seg = tid & 3;               // B tile: 64 rows x 32 bf16
      const uint4* gp = (const uint4*)(Bt + (long)(bnn + row) * KGEMM + k0 + seg * 8);
      *((uint4*)&sB[row][seg * 8]) = *gp;
    }
    __syncthreads();
    bf16x16 afrag[2], bfrag[2];
#pragma unroll
    for (int i = 0; i < 2; ++i) {
      union { bf16x16 v; unsigned int u[8]; } fa, fb;
      const unsigned int* pa = (const unsigned int*)&sA[wm + i * 16 + rrow][0];
      const unsigned int* pb = (const unsigned int*)&sB[wn + i * 16 + rrow][0];
#pragma unroll
      for (int q = 0; q < 4; ++q) {
        // ISA 7.12.2: lane half 0 holds K 0..7 & 16..23, half 1 holds K 8..15 & 24..31
        fa.u[q]     = pa[hsel * 4 + q];
        fa.u[4 + q] = pa[8 + hsel * 4 + q];
        fb.u[q]     = pb[hsel * 4 + q];
        fb.u[4 + q] = pb[8 + hsel * 4 + q];
      }
      afrag[i] = fa.v; bfrag[i] = fb.v;
    }
#pragma unroll
    for (int i = 0; i < 2; ++i)
#pragma unroll
      for (int j = 0; j < 2; ++j)
        acc[i][j] = __builtin_amdgcn_wmma_f32_16x16x32_bf16(
            false, afrag[i], false, bfrag[j], (short)0, acc[i][j], false, false);
    __syncthreads();
  }
  int rowoff = (lane >> 4) * 8;
  int coloff = lane & 15;
#pragma unroll
  for (int i = 0; i < 2; ++i)
#pragma unroll
    for (int j = 0; j < 2; ++j) {
      int ncol = bnn + wn + j * 16 + coloff;
      float win = 0.5f - 0.5f * __cosf(6.28318530717958647692f * (float)ncol * (1.f / 2048.f));
#pragma unroll
      for (int r = 0; r < 8; ++r) {
        int mrow = bm + wm + i * 16 + rowoff + r;
        frames[(long)mrow * NGEMM + ncol] = acc[i][j][r] * win;
      }
    }
}

// ---------------------------------------------------------------- stage 5: half-lapped overlap-add
__global__ void k_ola(const float* __restrict__ frames, float* __restrict__ res) {
  long idx = (long)blockIdx.x * 256 + threadIdx.x;   // BEDIM * NSAMPLES
  int be = (int)(idx >> 18);
  int s = (int)(idx & (NSAMPLES - 1));
  float v = frames[(long)(be * NFRAMES + (s >> 10)) * NGEMM + (s & 1023)];
  if (s >= STEP) {
    int s2 = s - STEP;
    v += frames[(long)(be * NFRAMES + (s2 >> 10)) * NGEMM + STEP + (s2 & 1023)];
  }
  res[idx] = v;
}

// noise signal: interp(env^2) * noise_u, zero-padded past 65536
__global__ void k_noise(const float* __restrict__ env2, const float* __restrict__ noise_u,
                        float* __restrict__ nsig) {
  long idx = (long)blockIdx.x * 256 + threadIdx.x;   // 32 * NSAMPLES
  int ne = (int)(idx >> 18);
  int s = (int)(idx & (NSAMPLES - 1));
  float v = 0.f;
  if (s < 65536) {
    float pos = ((float)s + 0.5f) * (64.f / 65536.f) - 0.5f;
    pos = fminf(fmaxf(pos, 0.f), 63.f);
    int lo = (int)pos; int hi = min(lo + 1, 63);
    float w = pos - (float)lo;
    float e = env2[ne * 64 + lo] * (1.f - w) + env2[ne * 64 + hi] * w;
    v = e * noise_u[idx];
  }
  nsig[idx] = v;
}

// ---------------------------------------------------------------- stage 6: FFT convolution (Stockham)
__global__ void k_fft_load(const float* __restrict__ res, const float* __restrict__ nsig,
                           float2* __restrict__ buf, int ev0) {
  int si = blockIdx.y;
  long s = (long)blockIdx.x * 256 + threadIdx.x;
  int el = si / 3, r = si % 3;
  int evn = ev0 + el;
  float v = 0.f;
  if (s < NSAMPLES)
    v = (r < 2) ? res[(long)(2 * evn + r) * NSAMPLES + s]
                : nsig[(long)evn * NSAMPLES + s];
  buf[(long)si * FFT_N + s] = make_float2(v, 0.f);
}

__global__ void k_pass4(const float2* __restrict__ src, float2* __restrict__ dst,
                        int Ns, float sgn) {
  int t = blockIdx.x * 256 + threadIdx.x;          // 0 .. FFT_N/4-1
  const int Nq = FFT_N / 4;
  const float2* s = src + (long)blockIdx.y * FFT_N;
  float2* d = dst + (long)blockIdx.y * FFT_N;
  int pos = t % Ns, grp = t / Ns;
  float2 v0 = s[t], v1 = s[t + Nq], v2 = s[t + 2 * Nq], v3 = s[t + 3 * Nq];
  float ang = sgn * 6.28318530717958647692f * (float)pos / (4.f * (float)Ns);  // |ang| < pi/2
  float si, co; __sincosf(ang, &si, &co);
  float2 w1 = make_float2(co, si);
  float2 w2 = cmul(w1, w1);
  float2 w3 = cmul(w2, w1);
  v1 = cmul(v1, w1); v2 = cmul(v2, w2); v3 = cmul(v3, w3);
  float2 t0 = make_float2(v0.x + v2.x, v0.y + v2.y);
  float2 t1 = make_float2(v0.x - v2.x, v0.y - v2.y);
  float2 t2 = make_float2(v1.x + v3.x, v1.y + v3.y);
  float2 t3 = make_float2(v1.x - v3.x, v1.y - v3.y);
  float2 jt = make_float2(-sgn * t3.y, sgn * t3.x);   // (i*sgn) * t3
  long base = (long)grp * 4 * Ns + pos;
  d[base]          = make_float2(t0.x + t2.x, t0.y + t2.y);
  d[base + Ns]     = make_float2(t1.x + jt.x, t1.y + jt.y);
  d[base + 2 * Ns] = make_float2(t0.x - t2.x, t0.y - t2.y);
  d[base + 3 * Ns] = make_float2(t1.x - jt.x, t1.y - jt.y);
}

__global__ void k_pass2(const float2* __restrict__ src, float2* __restrict__ dst,
                        int Ns, float sgn) {
  int t = blockIdx.x * 256 + threadIdx.x;          // 0 .. FFT_N/2-1
  const int Nh = FFT_N / 2;
  const float2* s = src + (long)blockIdx.y * FFT_N;
  float2* d = dst + (long)blockIdx.y * FFT_N;
  int pos = t % Ns, grp = t / Ns;
  float2 v0 = s[t], v1 = s[t + Nh];
  float ang = sgn * 6.28318530717958647692f * (float)pos / (2.f * (float)Ns);
  float si, co; __sincosf(ang, &si, &co);
  v1 = cmul(v1, make_float2(co, si));
  long base = (long)grp * 2 * Ns + pos;
  d[base]      = make_float2(v0.x + v1.x, v0.y + v1.y);
  d[base + Ns] = make_float2(v0.x - v1.x, v0.y - v1.y);
}

__global__ void k_specmul(float2* __restrict__ buf) {
  int el = blockIdx.y;                              // 0..CHUNK-1
  long s = (long)blockIdx.x * 256 + threadIdx.x;
  float2 nn = buf[(long)(el * 3 + 2) * FFT_N + s];
  for (int r = 0; r < 2; ++r) {
    long o = (long)(el * 3 + r) * FFT_N + s;
    buf[o] = cmul(buf[o], nn);
  }
}

__global__ void k_fft_store(const float2* __restrict__ buf, float* __restrict__ res, int ev0) {
  int y = blockIdx.y;                               // 0..2*CHUNK-1
  long s = (long)blockIdx.x * 256 + threadIdx.x;    // 0..NSAMPLES-1
  int el = y >> 1, r = y & 1;
  res[(long)(2 * (ev0 + el) + r) * NSAMPLES + s] =
      buf[(long)(el * 3 + r) * FFT_N + s].x * (1.f / (float)FFT_N);
}

// ---------------------------------------------------------------- stage 7: deformation * res, mix with noise
__global__ void k_mix(const float* __restrict__ res, const float* __restrict__ nsig,
                      const float* __restrict__ dc, const float* __restrict__ wmix,
                      float* __restrict__ out) {
  long idx = (long)blockIdx.x * 256 + threadIdx.x;  // 32 * NSAMPLES
  int ne = (int)(idx >> 18);
  int s = (int)(idx & (NSAMPLES - 1));
  float pos = ((float)s + 0.5f) * (256.f / 262144.f) - 0.5f;
  pos = fminf(fmaxf(pos, 0.f), 255.f);
  int lo = (int)pos; int hi = min(lo + 1, 255);
  float w = pos - (float)lo;
  float x = 0.f;
  for (int e = 0; e < 2; ++e) {
    int be = 2 * ne + e;
    float dv = dc[be * NFRAMES + lo] * (1.f - w) + dc[be * NFRAMES + hi] * w;
    x += res[(long)be * NSAMPLES + s] * dv;
  }
  out[idx] = wmix[ne * 2 + 0] * nsig[idx] + wmix[ne * 2 + 1] * x;
}

// ================================================================ host
extern "C" void kernel_launch(void* const* d_in, const int* in_sizes, int n_in,
                              void* d_out, int out_size, void* d_ws, size_t ws_size,
                              hipStream_t stream) {
  (void)in_sizes; (void)n_in; (void)out_size; (void)ws_size;
  const float* ev        = (const float*)d_in[0];
  const float* Wn        = (const float*)d_in[1];
  const float* bn        = (const float*)d_in[2];
  const float* items_n   = (const float*)d_in[3];
  const float* Wd        = (const float*)d_in[4];
  const float* bd        = (const float*)d_in[5];
  const float* items_d   = (const float*)d_in[6];
  const float* Wc        = (const float*)d_in[7];
  const float* bc        = (const float*)d_in[8];
  const float* items_c   = (const float*)d_in[9];
  const float* Wp        = (const float*)d_in[10];
  const float* bp        = (const float*)d_in[11];
  const float* items_p   = (const float*)d_in[12];
  const float* Wdef      = (const float*)d_in[13];
  const float* bdef      = (const float*)d_in[14];
  const float* items_def = (const float*)d_in[15];
  const float* Wamp      = (const float*)d_in[16];
  const float* bamp      = (const float*)d_in[17];
  const float* Wmix      = (const float*)d_in[18];
  const float* bmix      = (const float*)d_in[19];
  const float* noise_u   = (const float*)d_in[20];
  const float* dn        = (const float*)d_in[21];
  float* out = (float*)d_out;

  char* ws = (char*)d_ws;
  size_t off = 0;
  auto carve = [&](size_t bytes) -> void* {
    void* p = ws + off;
    off = (off + bytes + 255) & ~(size_t)255;
    return p;
  };
  float* hN    = (float*)carve(32 * 32 * 4);
  float* hD    = (float*)carve(32 * 32 * 4);
  float* hC    = (float*)carve(32 * 128 * 4);
  float* hP    = (float*)carve(32 * 16 * 4);
  float* hDef  = (float*)carve(32 * 32 * 4);
  float* wmix  = (float*)carve(32 * 2 * 4);
  float* env2  = (float*)carve(32 * 64 * 4);
  float* cumdn = (float*)carve((size_t)NFRAMES * NCOEFFS * 4);
  float* logc  = (float*)carve((size_t)BEDIM * NCOEFFS * 4);
  float* sp    = (float*)carve((size_t)BEDIM * NCOEFFS * 4);
  float* sm    = (float*)carve((size_t)BEDIM * NCOEFFS * 4);
  float* pd    = (float*)carve((size_t)BEDIM * NCOEFFS * 4);
  float* dc    = (float*)carve((size_t)BEDIM * NFRAMES * 4);
  unsigned short* Bt = (unsigned short*)carve((size_t)NGEMM * KGEMM * 2);
  unsigned short* Am = (unsigned short*)carve((size_t)MGEMM * KGEMM * 2);
  float* frames = (float*)carve((size_t)MGEMM * NGEMM * 4);
  float* res    = (float*)carve((size_t)BEDIM * NSAMPLES * 4);
  float* nsig   = (float*)carve((size_t)32 * NSAMPLES * 4);
  float2* fftA  = (float2*)carve((size_t)SIGS * FFT_N * 8);
  float2* fftB  = (float2*)carve((size_t)SIGS * FFT_N * 8);

  k_small<<<32, 256, 0, stream>>>(ev, Wn, bn, items_n, Wd, bd, Wc, bc, Wp, bp,
                                  Wdef, bdef, Wmix, bmix,
                                  hN, hD, hC, hP, hDef, wmix, env2);
  k_cumdn<<<(NCOEFFS + 255) / 256, 256, 0, stream>>>(dn, cumdn);
  k_params<<<(BEDIM * NCOEFFS + 255) / 256, 256, 0, stream>>>(
      ev, Wamp, bamp, items_d, items_c, items_p, hD, hC, hP, logc, sp, sm, pd);
  k_defo<<<BEDIM, 256, 0, stream>>>(hDef, items_def, dc);
  k_basis<<<(int)(((long)NGEMM * KGEMM) / 256), 256, 0, stream>>>(Bt);
  k_spec<<<(int)(((long)MGEMM * 1056) / 256), 256, 0, stream>>>(logc, sp, sm, pd, cumdn, Am);
  k_gemm<<<dim3(MGEMM / 128, NGEMM / 64), 256, 0, stream>>>(Am, Bt, frames);
  k_ola<<<(int)(((long)BEDIM * NSAMPLES) / 256), 256, 0, stream>>>(frames, res);
  k_noise<<<(int)(((long)32 * NSAMPLES) / 256), 256, 0, stream>>>(env2, noise_u, nsig);

  for (int chunk = 0; chunk < NEVENTS / CHUNK; ++chunk) {
    int ev0 = chunk * CHUNK;
    k_fft_load<<<dim3(FFT_N / 256, SIGS), 256, 0, stream>>>(res, nsig, fftA, ev0);
    float2* cur = fftA; float2* alt = fftB;
    int Ns = 1;
    for (int p = 0; p < 9; ++p) {                     // forward: 9 x radix-4
      k_pass4<<<dim3(FFT_N / 4 / 256, SIGS), 256, 0, stream>>>(cur, alt, Ns, -1.f);
      float2* tmp = cur; cur = alt; alt = tmp; Ns *= 4;
    }
    k_pass2<<<dim3(FFT_N / 2 / 256, SIGS), 256, 0, stream>>>(cur, alt, Ns, -1.f);
    { float2* tmp = cur; cur = alt; alt = tmp; }
    k_specmul<<<dim3(FFT_N / 256, CHUNK), 256, 0, stream>>>(cur);
    Ns = 1;
    for (int p = 0; p < 9; ++p) {                     // inverse: 9 x radix-4
      k_pass4<<<dim3(FFT_N / 4 / 256, SIGS), 256, 0, stream>>>(cur, alt, Ns, +1.f);
      float2* tmp = cur; cur = alt; alt = tmp; Ns *= 4;
    }
    k_pass2<<<dim3(FFT_N / 2 / 256, SIGS), 256, 0, stream>>>(cur, alt, Ns, +1.f);
    { float2* tmp = cur; cur = alt; alt = tmp; }
    k_fft_store<<<dim3(NSAMPLES / 256, 2 * CHUNK), 256, 0, stream>>>(cur, res, ev0);
  }

  k_mix<<<(int)(((long)32 * NSAMPLES) / 256), 256, 0, stream>>>(res, nsig, dc, wmix, out);
}
